// MultiModalModel_5592047419545
// MI455X (gfx1250) — matrix operand order repaired and checked by reading, not compiled, not run on hardware
//
#include <hip/hip_runtime.h>

// ---- CDNA5 (gfx1250, wave32) fused max-sim CLIP head -----------------------
// Kernel 1: gather embedding[text] once -> bf16 hi/lo in WMMA-A order (d_ws)
// Kernel 2: per workgroup = one image x all 6400 text words
//   GEMM  : M=16 words/tile x N=16 patches/tile x K=128, bf16x3 split WMMA
//   reduce: max over 49 patches -> sum over 25 words -> scale/len -> [I,T]+[T,I]

typedef __bf16 v16bf __attribute__((ext_vector_type(16)));
typedef __bf16 v8bf  __attribute__((ext_vector_type(8)));
typedef float  v8f   __attribute__((ext_vector_type(8)));
typedef float  v4f   __attribute__((ext_vector_type(4)));
typedef unsigned int v4u __attribute__((ext_vector_type(4)));
typedef int    v8i   __attribute__((ext_vector_type(8)));
typedef int    v4i   __attribute__((ext_vector_type(4)));

#define NI 256
#define NT 256
#define NL 25
#define NE 128
#define NP 49          // H*W patches
#define NPAD 64        // patches padded to 4 N-tiles of 16
#define BSTRIDE 136    // bf16 elems per LDS row: 128 + 8 pad = 272B (16B-aligned rows)
#define NWORDS (NT*NL) // 6400
#define NMT (NWORDS/16)// 400 M-tiles of 16 words
#define FLT_BIG 3.402823466e38f

__device__ __forceinline__ v16bf cat8(v8bf a, v8bf b) {
  return __builtin_shufflevector(a, b, 0,1,2,3,4,5,6,7,8,9,10,11,12,13,14,15);
}

// ---------------------------------------------------------------------------
// Prologue: text_features = embedding[text], split f32 -> bf16 hi + bf16 lo,
// stored permuted so a lane's v16bf A fragment is one contiguous 32B run:
//   dst[gw][kc*32 + 16h + e] = K = kc*32 + 8h + {0..7} , kc*32 + 16 + 8h + {0..7}
// ---------------------------------------------------------------------------
__global__ __launch_bounds__(256)
void pack_text_kernel(const float* __restrict__ emb,
                      const int*   __restrict__ text,
                      __bf16* __restrict__ wsAh,
                      __bf16* __restrict__ wsAl)
{
  int idx = blockIdx.x * 256 + threadIdx.x;     // NWORDS*8 work items
  if (idx >= NWORDS * 8) return;
  int gw  = idx >> 3;
  int seg = idx & 7;
  int kc  = seg >> 1;
  int h   = seg & 1;
  int t = gw / NL;
  int l = gw - t * NL;
  int token = text[t * NL + l];
  const float* er = emb + (size_t)token * NE;

  int k0 = kc * 32 + 8 * h;
  v4f f0 = *(const v4f*)(er + k0);
  v4f f1 = *(const v4f*)(er + k0 + 4);
  v4f f2 = *(const v4f*)(er + k0 + 16);
  v4f f3 = *(const v4f*)(er + k0 + 20);

  v8bf h0, h1, l0, l1;
#pragma unroll
  for (int j = 0; j < 4; ++j) {
    float x0 = f0[j], x1 = f1[j], x2 = f2[j], x3 = f3[j];
    __bf16 b0 = (__bf16)x0, b1 = (__bf16)x1, b2 = (__bf16)x2, b3 = (__bf16)x3;
    h0[j]     = b0; h0[4 + j] = b1;
    h1[j]     = b2; h1[4 + j] = b3;
    l0[j]     = (__bf16)(x0 - (float)b0);
    l0[4 + j] = (__bf16)(x1 - (float)b1);
    l1[j]     = (__bf16)(x2 - (float)b2);
    l1[4 + j] = (__bf16)(x3 - (float)b3);
  }
  size_t base = (size_t)gw * NE + kc * 32 + 16 * h;
  *(v8bf*)(wsAh + base)     = h0;
  *(v8bf*)(wsAh + base + 8) = h1;
  *(v8bf*)(wsAl + base)     = l0;
  *(v8bf*)(wsAl + base + 8) = l1;
}

// ---------------------------------------------------------------------------
// Main kernel. FROMWS=true loads pre-split bf16 A fragments from d_ws;
// FROMWS=false gathers+converts from the f32 embedding (fallback).
// ---------------------------------------------------------------------------
template <bool FROMWS>
__global__ __launch_bounds__(256)
void clip_maxsim_kernel(const float* __restrict__ img,    // [I,E,H,W]
                        const float* __restrict__ emb,    // [V,E]
                        const float* __restrict__ lscale, // [1]
                        const int*   __restrict__ text,   // [T,L]
                        const int*   __restrict__ tlen,   // [T]
                        const __bf16* __restrict__ wsAh,  // [NWORDS,NE] permuted
                        const __bf16* __restrict__ wsAl,  // [NWORDS,NE] permuted
                        float* __restrict__ out)          // [I,T] ++ [T,I]
{
  __shared__ __align__(16) __bf16 sBhi[NPAD * BSTRIDE]; // image tile, bf16 hi
  __shared__ __align__(16) __bf16 sBlo[NPAD * BSTRIDE]; // image tile, bf16 lo
  __shared__ __align__(16) float  sImg[NE * NP];        // TDM staging (f32)
  __shared__ float sSums[NT];                           // per-text running sums

  const int tid = threadIdx.x;
  const int i   = blockIdx.x;
  const float* gimg = img + (size_t)i * (NE * NP);

  sSums[tid] = 0.0f;

  // ---- stage image tile [E=128][HW=49] f32 into LDS via Tensor Data Mover --
#if defined(__HIP_DEVICE_COMPILE__) && __has_builtin(__builtin_amdgcn_tensor_load_to_lds)
  if (tid == 0) {
    unsigned long long ga = (unsigned long long)(const void*)gimg;
    unsigned lds = (unsigned)(unsigned long long)(const void*)sImg;
    v4u g0;
    g0.x = 1u;                                                // count=1 (valid D#)
    g0.y = lds;                                               // lds_addr
    g0.z = (unsigned)ga;                                      // global_addr lo
    g0.w = ((unsigned)(ga >> 32) & 0x01FFFFFFu) | (2u << 30); // addr hi | type=2
    v8i g1;
    g1[0] = (int)(2u << 16);           // data_size=2 (4B)
    g1[1] = (int)((unsigned)NP << 16); // tensor_dim0 = 49
    g1[2] = (int)((unsigned)NE << 16); // tensor_dim1 = 128
    g1[3] = (int)((unsigned)NP << 16); // tile_dim0 = 49
    g1[4] = NE;                        // tile_dim1 = 128, tile_dim2 = 0
    g1[5] = NP;                        // tensor_dim0_stride = 49
    g1[6] = 0;
    g1[7] = 0;
    v4i gz = {0, 0, 0, 0};
#if __clang_major__ >= 23
    v8i gz8 = {0, 0, 0, 0, 0, 0, 0, 0};
    __builtin_amdgcn_tensor_load_to_lds(g0, g1, gz, gz, gz8, 0);
#else
    __builtin_amdgcn_tensor_load_to_lds(g0, g1, gz, gz, 0);
#endif
    __builtin_amdgcn_s_wait_tensorcnt(0);
  }
  const float* srcImg = sImg;
#else
  const float* srcImg = gimg;
#endif
  __syncthreads();

  // ---- transpose e-major -> [patch][K], split f32 -> bf16 hi + bf16 lo ----
  for (int idx = tid; idx < NPAD * NE; idx += 256) {
    int n = idx & (NPAD - 1);
    int k = idx >> 6;
    float x = (n < NP) ? srcImg[k * NP + n] : 0.0f;
    __bf16 hh = (__bf16)x;
    float  rr = x - (float)hh;
    sBhi[n * BSTRIDE + k] = hh;
    sBlo[n * BSTRIDE + k] = (__bf16)rr;
  }
  __syncthreads();

  const int wave = tid >> 5;
  const int lane = tid & 31;
  const int half = lane >> 4;   // K-half selector for 16-bit A/B layouts
  const int sub  = lane & 15;   // A row (word) / B col (patch) within tile

  for (int mt = wave; mt < NMT; mt += 8) {
    int gw = mt * 16 + sub;
    v16bf Ah[4], Al[4];

    if constexpr (FROMWS) {
      // pre-split, pre-permuted: one 32B contiguous run per (kc, matrix)
      const __bf16* ah = wsAh + (size_t)gw * NE + 16 * half;
      const __bf16* al = wsAl + (size_t)gw * NE + 16 * half;
#pragma unroll
      for (int kc = 0; kc < 4; ++kc) {
        Ah[kc] = cat8(*(const v8bf*)(ah + kc * 32), *(const v8bf*)(ah + kc * 32 + 8));
        Al[kc] = cat8(*(const v8bf*)(al + kc * 32), *(const v8bf*)(al + kc * 32 + 8));
      }
    } else {
      int t = gw / NL;
      int l = gw - t * NL;
      int token = text[t * NL + l];
      const float* er = emb + (size_t)token * NE;
#pragma unroll
      for (int kc = 0; kc < 4; ++kc) {
        int k0 = kc * 32 + half * 8;
        v4f f0 = *(const v4f*)(er + k0);
        v4f f1 = *(const v4f*)(er + k0 + 4);
        v4f f2 = *(const v4f*)(er + k0 + 16);
        v4f f3 = *(const v4f*)(er + k0 + 20);
#pragma unroll
        for (int j = 0; j < 4; ++j) {
          float x0 = f0[j], x1 = f1[j], x2 = f2[j], x3 = f3[j];
          __bf16 b0 = (__bf16)x0, b1 = (__bf16)x1, b2 = (__bf16)x2, b3 = (__bf16)x3;
          Ah[kc][j]      = b0; Ah[kc][4 + j]  = b1;
          Ah[kc][8 + j]  = b2; Ah[kc][12 + j] = b3;
          Al[kc][j]      = (__bf16)(x0 - (float)b0);
          Al[kc][4 + j]  = (__bf16)(x1 - (float)b1);
          Al[kc][8 + j]  = (__bf16)(x2 - (float)b2);
          Al[kc][12 + j] = (__bf16)(x3 - (float)b3);
        }
      }
    }

    v8f zero = {0.f, 0.f, 0.f, 0.f, 0.f, 0.f, 0.f, 0.f};
    v8f acc[4];
#pragma unroll
    for (int nt = 0; nt < 4; ++nt) acc[nt] = zero;

    // ---- bf16x3 WMMA: D += Ah*Bh + Ah*Bl + Al*Bh, 4 K-chunks, 4 N-tiles ----
#pragma unroll
    for (int kc = 0; kc < 4; ++kc) {
      int kb = kc * 32 + half * 8;              // B layout: runs of 8 at kb and kb+16
#pragma unroll
      for (int nt = 0; nt < 4; ++nt) {
        const __bf16* bp = &sBhi[(nt * 16 + sub) * BSTRIDE + kb];
        const __bf16* lp = &sBlo[(nt * 16 + sub) * BSTRIDE + kb];
        v16bf Bh = cat8(*(const v8bf*)bp, *(const v8bf*)(bp + 16));
        v16bf Bl = cat8(*(const v8bf*)lp, *(const v8bf*)(lp + 16));
        acc[nt] = __builtin_amdgcn_wmma_f32_16x16x32_bf16(
            false, Ah[kc], false, Bh, (short)0, acc[nt], false, false);
        acc[nt] = __builtin_amdgcn_wmma_f32_16x16x32_bf16(
            false, Ah[kc], false, Bl, (short)0, acc[nt], false, false);
        acc[nt] = __builtin_amdgcn_wmma_f32_16x16x32_bf16(
            false, Al[kc], false, Bh, (short)0, acc[nt], false, false);
      }
    }

    // ---- max over patches: mask pad cols, fold 4 N-tiles, lane butterfly ----
    float rmax[8];
#pragma unroll
    for (int r = 0; r < 8; ++r) rmax[r] = -FLT_BIG;
#pragma unroll
    for (int nt = 0; nt < 4; ++nt) {
      bool valid = (nt * 16 + sub) < NP;
#pragma unroll
      for (int r = 0; r < 8; ++r) {
        float x = valid ? acc[nt][r] : -FLT_BIG;
        rmax[r] = fmaxf(rmax[r], x);
      }
    }
#pragma unroll
    for (int s = 1; s <= 8; s <<= 1) {
#pragma unroll
      for (int r = 0; r < 8; ++r)
        rmax[r] = fmaxf(rmax[r], __shfl_xor(rmax[r], s, 32));
    }

    // C layout: VGPR r holds M=r (lanes 0-15) / M=8+r (lanes 16-31)
    if (sub == 0) {
      int rowbase = mt * 16 + half * 8;
#pragma unroll
      for (int r = 0; r < 8; ++r) {
        int tt = (rowbase + r) / NL;
        atomicAdd(&sSums[tt], rmax[r]);        // ds_add_f32, sum over words l
      }
    }
  }

  __syncthreads();

  // ---- epilogue: scale / text_length, write [I,T] and its transpose --------
  float scale = expf(lscale[0]);
  float s = sSums[tid] * scale / (float)tlen[tid];
  out[(size_t)i * NT + tid] = s;                       // logits_per_image
  out[(size_t)NI * NT + (size_t)tid * NI + i] = s;     // logits_per_text
}

extern "C" void kernel_launch(void* const* d_in, const int* in_sizes, int n_in,
                              void* d_out, int out_size, void* d_ws, size_t ws_size,
                              hipStream_t stream) {
  const float* img  = (const float*)d_in[0];   // image_features [I,E,H,W] f32
  const float* emb  = (const float*)d_in[1];   // embedding [V,E] f32
  const float* ls   = (const float*)d_in[2];   // logit_scale [1] f32
  const int*   text = (const int*)d_in[3];     // text [T,L] i32
  const int*   tlen = (const int*)d_in[4];     // text_length [T] i32
  float* out = (float*)d_out;                  // [I,T] ++ [T,I] f32

  const size_t needA = (size_t)NWORDS * NE * sizeof(__bf16);  // 1.64 MB each
  __bf16* wsAh = (__bf16*)d_ws;
  __bf16* wsAl = wsAh + (size_t)NWORDS * NE;

  if (ws_size >= 2 * needA) {
    pack_text_kernel<<<dim3((NWORDS * 8 + 255) / 256), dim3(256), 0, stream>>>(
        emb, text, wsAh, wsAl);
    clip_maxsim_kernel<true><<<dim3(NI), dim3(256), 0, stream>>>(
        img, emb, ls, text, tlen, wsAh, wsAl, out);
  } else {
    clip_maxsim_kernel<false><<<dim3(NI), dim3(256), 0, stream>>>(
        img, emb, ls, text, tlen, nullptr, nullptr, out);
  }
  (void)in_sizes; (void)n_in; (void)out_size;
}
